// MultiHeadSelfAttention_7129645711892
// MI455X (gfx1250) — matrix operand order repaired and checked by reading, not compiled, not run on hardware
//
#include <hip/hip_runtime.h>

typedef __bf16 bf16;
typedef __attribute__((ext_vector_type(16))) __bf16 v16bf;
typedef __attribute__((ext_vector_type(8)))  __bf16 v8bf;
typedef __attribute__((ext_vector_type(4)))  __bf16 v4bf;
typedef __attribute__((ext_vector_type(8)))  float   v8f;
typedef __attribute__((ext_vector_type(4)))  unsigned int v4u;
typedef __attribute__((ext_vector_type(8)))  int     v8i;
typedef __attribute__((ext_vector_type(4)))  int     v4i;

#if defined(__has_builtin)
#if __has_builtin(__builtin_amdgcn_tensor_load_to_lds) && \
    __has_builtin(__builtin_amdgcn_s_wait_tensorcnt)
#define HAVE_TDM 1
#endif
#endif
#ifndef HAVE_TDM
#define HAVE_TDM 0
#endif

// ---------------------------------------------------------------------------
// WMMA helper: D = A(16x32 bf16) x B(32x16 bf16) + C(16x16 f32)
// ---------------------------------------------------------------------------
__device__ __forceinline__ v8f wmma_bf16(v16bf a, v16bf b, v8f c) {
  return __builtin_amdgcn_wmma_f32_16x16x32_bf16(
      /*neg_a=*/false, a, /*neg_b=*/false, b,
      /*c_mod=*/(short)0, c, /*reuse_a=*/false, /*reuse_b=*/false);
}

// A fragment (16x32), source tile row-major [M][K], K contiguous, stride ld.
__device__ __forceinline__ v16bf load_frag_a(const bf16* p, int ld) {
  const int lane = threadIdx.x & 31;
  const int row  = lane & 15;
  const int kb   = (lane >> 4) << 3;
  const bf16* r = p + row * ld;
  v16bf a;
#pragma unroll
  for (int e = 0; e < 8; ++e) a[e] = r[kb + e];
#pragma unroll
  for (int e = 0; e < 8; ++e) a[8 + e] = r[16 + kb + e];
  return a;
}

// B fragment (32x16), source tile [N][K] (K contiguous), stride ld.
__device__ __forceinline__ v16bf load_frag_b(const bf16* p, int ld) {
  const int lane = threadIdx.x & 31;
  const int n    = lane & 15;
  const int kb   = (lane >> 4) << 3;
  const bf16* r = p + n * ld + kb;
  v16bf b;
#pragma unroll
  for (int e = 0; e < 8; ++e) b[e] = r[e];
#pragma unroll
  for (int e = 0; e < 8; ++e) b[8 + e] = r[16 + e];
  return b;
}

__device__ __forceinline__ v4bf cvt4(const float* p) {
  const float4 f = *(const float4*)p;
  v4bf r; r[0] = (bf16)f.x; r[1] = (bf16)f.y; r[2] = (bf16)f.z; r[3] = (bf16)f.w;
  return r;
}
__device__ __forceinline__ v4bf cvt4(const bf16* p) { return *(const v4bf*)p; }

// 16-lane max reduction via DPP16 butterfly (masks xor1, xor2, xor7, xor15).
template <int CTRL>
__device__ __forceinline__ float fmax_dpp(float x) {
  const int y = __builtin_amdgcn_mov_dpp(__builtin_bit_cast(int, x),
                                         CTRL, 0xf, 0xf, true);
  return fmaxf(x, __builtin_bit_cast(float, y));
}
__device__ __forceinline__ float rowmax16(float x) {
  x = fmax_dpp<0x0B1>(x);  // quad_perm(1,0,3,2)  : xor 1
  x = fmax_dpp<0x04E>(x);  // quad_perm(2,3,0,1)  : xor 2
  x = fmax_dpp<0x141>(x);  // row_half_mirror     : xor 7
  x = fmax_dpp<0x140>(x);  // row_mirror          : xor 15
  return x;
}

#if HAVE_TDM
// ---------------------------------------------------------------------------
// TDM: DMA a 64x64 bf16 tile (row stride 64 elems in memory) into LDS with
// hardware row padding 128B->144B (matches LDK=72). D# per ISA 08 §8.3/8.4.
// ---------------------------------------------------------------------------
__device__ __forceinline__ void tdm_load_tile_64x64_bf16(const bf16* gsrc,
                                                         unsigned lds_off) {
  const unsigned long long ga = (unsigned long long)(uintptr_t)gsrc;
  v4u g0;
  g0[0] = 1u;                                    // count=1 valid user descriptor
  g0[1] = lds_off;                               // lds_addr (bytes)
  g0[2] = (unsigned)ga;                          // global_addr[31:0]
  g0[3] = (unsigned)((ga >> 32) & 0x01FFFFFFu)   // global_addr[56:32]
        | (2u << 30);                            // type = 2 ("image")
  v8i g1;
  g1[0] = (int)((1u << 16)    // data_size = 2 bytes
              | (1u << 20)    // pad_enable
              | (4u << 22)    // pad_interval = 32 DWORDs (one 64xbf16 row)
              | (3u << 25));  // pad_amount   = 4 DWORDs (8 bf16)
  g1[1] = (int)(64u << 16);   // tensor_dim0 = 64 (lo16)
  g1[2] = (int)(64u << 16);   // tensor_dim0 hi = 0 | tensor_dim1 = 64 (lo16)
  g1[3] = (int)(64u << 16);   // tensor_dim1 hi = 0 | tile_dim0 = 64
  g1[4] = (int)64u;           // tile_dim1 = 64 | tile_dim2 = 0
  g1[5] = (int)64u;           // tensor_dim0_stride = 64 (lo32)
  g1[6] = 0;                  // stride0 hi16 = 0 | stride1 lo16 = 0
  g1[7] = 0;                  // stride1 hi32 = 0
  const v4i gz = {0, 0, 0, 0};
#if __clang_major__ >= 23
  const v8i gz8 = {0, 0, 0, 0, 0, 0, 0, 0};
  __builtin_amdgcn_tensor_load_to_lds(g0, g1, gz, gz, gz8, 0);
#else
  __builtin_amdgcn_tensor_load_to_lds(g0, g1, gz, gz, 0);
#endif
}
#endif  // HAVE_TDM

// ---------------------------------------------------------------------------
// Tiled WMMA GEMM:  out[M,Nout] = A[M,K] * W[Nout,K]^T + bias
// Block tile 128x128, 8 waves in 2x4 grid, 64x32 per wave (8 WMMA / k-step).
// EPI==0: QKV epilogue -> scatter bf16 q/k/v in [B,H,N,D], q scaled by 1/8
// EPI==1: f32 epilogue -> outF[M,Nout]
// ---------------------------------------------------------------------------
constexpr int BM = 128, BN = 128, BK = 32, LDT = 40;  // 40 = 32+8 pad

template <typename AT, int EPI>
__global__ __launch_bounds__(256) void gemm_bf16_wmma(
    const AT* __restrict__ A, const float* __restrict__ W,
    const float* __restrict__ bias, float* __restrict__ outF,
    bf16* __restrict__ oq, bf16* __restrict__ ok, bf16* __restrict__ ov,
    int M, int Nout, int K) {
  __shared__ bf16 aT[BM * LDT];
  __shared__ bf16 bT[BN * LDT];
  const int tid  = threadIdx.x;
  const int wave = tid >> 5, lane = tid & 31;
  const int wm = wave & 1, wn = wave >> 1;            // 2x4 wave grid
  const int m0 = blockIdx.y * BM, n0 = blockIdx.x * BN;

  v8f acc[4][2] = {};

  for (int kt = 0; kt < K; kt += BK) {
    // stage A tile 128x32 and B tile 128x32 (4+4 vec4 per thread)
#pragma unroll
    for (int i = 0; i < 4; ++i) {
      const int idx = tid + (i << 8);
      const int row = idx >> 3, col = (idx & 7) << 2;
      const AT*    ap = A + (size_t)(m0 + row) * K + kt + col;
      const float* wp = W + (size_t)(n0 + row) * K + kt + col;
      *(v4bf*)&aT[row * LDT + col] = cvt4(ap);
      *(v4bf*)&bT[row * LDT + col] = cvt4(wp);
      if (kt + BK < K) {                // speculative L2 warm-up for next k-step
        __builtin_prefetch(ap + BK, 0, 1);
        __builtin_prefetch(wp + BK, 0, 1);
      }
    }
    __syncthreads();

    v16bf af[4], bfr[2];
#pragma unroll
    for (int i = 0; i < 4; ++i)
      af[i] = load_frag_a(&aT[(wm * 64 + i * 16) * LDT], LDT);
#pragma unroll
    for (int j = 0; j < 2; ++j)
      bfr[j] = load_frag_b(&bT[(wn * 32 + j * 16) * LDT], LDT);
#pragma unroll
    for (int i = 0; i < 4; ++i)
#pragma unroll
      for (int j = 0; j < 2; ++j)
        acc[i][j] = wmma_bf16(af[i], bfr[j], acc[i][j]);
    __syncthreads();
  }

  const int lrow = lane & 15, rsel = lane >> 4;
#pragma unroll
  for (int i = 0; i < 4; ++i)
#pragma unroll
    for (int j = 0; j < 2; ++j)
#pragma unroll
      for (int vv = 0; vv < 8; ++vv) {
        const int mg = m0 + wm * 64 + i * 16 + vv + (rsel << 3);
        const int ng = n0 + wn * 32 + j * 16 + lrow;
        float val = acc[i][j][vv] + bias[ng];
        if constexpr (EPI == 1) {
          outF[(size_t)mg * Nout + ng] = val;
        } else {
          const int sec = ng >> 10, c = ng & 1023;
          const int h = c >> 6, d = c & 63;
          const int bb = mg >> 11, row = mg & 2047;
          if (sec == 0) val *= 0.125f;  // 1/sqrt(64)
          bf16* dst = (sec == 0) ? oq : (sec == 1) ? ok : ov;
          dst[((((size_t)bb * 16 + h) * 2048) + row) * 64 + d] = (bf16)val;
        }
      }
}

// ---------------------------------------------------------------------------
// Flash attention: per (b,h); 128 query rows / block, 16 / wave; key tiles of 64.
// K tiles staged by the Tensor Data Mover (double-buffered, TENSORcnt-tracked);
// V transposed at staging; row-max via DPP; row-sum via P x ones WMMA.
// ---------------------------------------------------------------------------
__global__ __launch_bounds__(256) void attn_flash_wmma(
    const bf16* __restrict__ q, const bf16* __restrict__ k,
    const bf16* __restrict__ v, bf16* __restrict__ o) {
  constexpr int LDK = 72;                 // 64 + 8 pad (=144B rows, 16B aligned)
  __shared__ bf16 kTbuf[2][64 * LDK];     // [key][d], TDM double buffer
  __shared__ bf16 vTt[64 * LDK];          // [d][key]  (transposed at staging)
  __shared__ bf16 pT[8 * 16 * LDK];       // per-wave P staging
  const int tid = threadIdx.x, wave = tid >> 5, lane = tid & 31;
  const int lrow = lane & 15, rsel = lane >> 4;
  const int bh = blockIdx.y;
  const int bb = bh >> 4, h = bh & 15;
  const size_t head = (size_t)bh * 2048 * 64;
  const int q0 = blockIdx.x * 128 + wave * 16;

  // Q fragments: 16 rows x 64 d -> two 16x32 A-frags, loaded straight from HBM
  v16bf qf[2];
  {
    const bf16* qr = q + head + (size_t)(q0 + lrow) * 64;
    const int kb8 = rsel << 3;
#pragma unroll
    for (int f = 0; f < 2; ++f) {
      const v8bf lo = *(const v8bf*)(qr + f * 32 + kb8);
      const v8bf hi = *(const v8bf*)(qr + f * 32 + 16 + kb8);
      v16bf t;
#pragma unroll
      for (int e = 0; e < 8; ++e) { t[e] = lo[e]; t[8 + e] = hi[e]; }
      qf[f] = t;
    }
  }

  v16bf onesb;                            // all-ones B fragment (layout-invariant)
#pragma unroll
  for (int e = 0; e < 16; ++e) onesb[e] = (bf16)1.0f;

  float mrun[8], lrun[8];
  v8f of[4] = {};
#pragma unroll
  for (int vv = 0; vv < 8; ++vv) { mrun[vv] = -3.0e38f; lrun[vv] = 0.0f; }

#if HAVE_TDM
  if (wave == 0)  // prologue: DMA first K tile
    tdm_load_tile_64x64_bf16(k + head, (unsigned)(uintptr_t)&kTbuf[0][0]);
#endif

  for (int it = 0; it < 32; ++it) {
    const int kt0 = it << 6;
    const bf16* kcur = &kTbuf[it & 1][0];

    // stage V (transposed); prefetch next V tile
#pragma unroll
    for (int i = 0; i < 2; ++i) {
      const int c = tid + (i << 8);
      const int row = c >> 3, off = (c & 7) << 3;   // key row, d offset
      const bf16* vsrc = v + head + (size_t)(kt0 + row) * 64 + off;
      const v8bf vv8 = *(const v8bf*)vsrc;
      if (kt0 + 64 < 2048) __builtin_prefetch(vsrc + 64 * 64, 0, 1);
#pragma unroll
      for (int e = 0; e < 8; ++e) vTt[(off + e) * LDK + row] = vv8[e];
    }
#if HAVE_TDM
    if (wave == 0) __builtin_amdgcn_s_wait_tensorcnt(0);  // K tile `it` in LDS
#else
#pragma unroll
    for (int i = 0; i < 2; ++i) {
      const int c = tid + (i << 8);
      const int row = c >> 3, off = (c & 7) << 3;
      *(v8bf*)&kTbuf[it & 1][row * LDK + off] =
          *(const v8bf*)(k + head + (size_t)(kt0 + row) * 64 + off);
    }
#endif
    __syncthreads();
#if HAVE_TDM
    if (wave == 0 && it + 1 < 32)  // overlap next K-tile DMA with this compute
      tdm_load_tile_64x64_bf16(k + head + (size_t)(kt0 + 64) * 64,
                               (unsigned)(uintptr_t)&kTbuf[(it + 1) & 1][0]);
#endif

    // S = Q * K^T   (16 x 64 per wave; 8 WMMAs)
    v8f sf[4];
#pragma unroll
    for (int j = 0; j < 4; ++j) {
      v8f c = {};
      c = wmma_bf16(qf[0], load_frag_b(&kcur[(j * 16) * LDK + 0 ], LDK), c);
      c = wmma_bf16(qf[1], load_frag_b(&kcur[(j * 16) * LDK + 32], LDK), c);
      sf[j] = c;
    }

    // online softmax: row-max via DPP butterfly (no LDS traffic)
    float alpha[8];
#pragma unroll
    for (int vv = 0; vv < 8; ++vv) {
      float mv = sf[0][vv];
#pragma unroll
      for (int j = 1; j < 4; ++j) mv = fmaxf(mv, sf[j][vv]);
      mv = rowmax16(mv);
      const float mn = fmaxf(mrun[vv], mv);
      alpha[vv] = __expf(mrun[vv] - mn);
      mrun[vv] = mn;
    }
#pragma unroll
    for (int j = 0; j < 4; ++j)
#pragma unroll
      for (int vv = 0; vv < 8; ++vv)
        sf[j][vv] = __expf(sf[j][vv] - mrun[vv]);
#pragma unroll
    for (int jd = 0; jd < 4; ++jd)
#pragma unroll
      for (int vv = 0; vv < 8; ++vv) of[jd][vv] = of[jd][vv] * alpha[vv];

    // C-layout f32 P -> A-layout bf16 P via per-wave LDS tile
    bf16* pw = &pT[wave * 16 * LDK];
#pragma unroll
    for (int j = 0; j < 4; ++j)
#pragma unroll
      for (int vv = 0; vv < 8; ++vv)
        pw[(vv + (rsel << 3)) * LDK + j * 16 + lrow] = (bf16)sf[j][vv];
    asm volatile("s_wait_dscnt 0x0" ::: "memory");  // LDS RAW fence (same wave)

    const v16bf pf0 = load_frag_a(pw, LDK);       // keys 0..31
    const v16bf pf1 = load_frag_a(pw + 32, LDK);  // keys 32..63

    // row-sum of P on the matrix unit: P x ones -> every column = rowsum
    {
      v8f sacc = {};
      sacc = wmma_bf16(pf0, onesb, sacc);
      sacc = wmma_bf16(pf1, onesb, sacc);
#pragma unroll
      for (int vv = 0; vv < 8; ++vv)
        lrun[vv] = lrun[vv] * alpha[vv] + sacc[vv];
    }

    // O += P * V   (contiguous B-frags from transposed V tile)
#pragma unroll
    for (int jd = 0; jd < 4; ++jd) {
      of[jd] = wmma_bf16(pf0, load_frag_b(&vTt[(jd * 16) * LDK + 0 ], LDK), of[jd]);
      of[jd] = wmma_bf16(pf1, load_frag_b(&vTt[(jd * 16) * LDK + 32], LDK), of[jd]);
    }
    __syncthreads();
  }

  // normalize and emit bf16 [B*N, C]
#pragma unroll
  for (int vv = 0; vv < 8; ++vv) {
    const float inv = 1.0f / lrun[vv];
    const size_t grow = (size_t)bb * 2048 + q0 + vv + (rsel << 3);
#pragma unroll
    for (int jd = 0; jd < 4; ++jd)
      o[grow * 1024 + h * 64 + jd * 16 + lrow] = (bf16)(of[jd][vv] * inv);
  }
}

// ---------------------------------------------------------------------------
extern "C" void kernel_launch(void* const* d_in, const int* in_sizes, int n_in,
                              void* d_out, int out_size, void* d_ws, size_t ws_size,
                              hipStream_t stream) {
  (void)in_sizes; (void)n_in; (void)out_size; (void)ws_size;
  const float* x     = (const float*)d_in[0];
  const float* qkv_w = (const float*)d_in[1];
  const float* qkv_b = (const float*)d_in[2];
  const float* fc_w  = (const float*)d_in[3];
  const float* fc_b  = (const float*)d_in[4];
  float* out = (float*)d_out;

  const size_t T = (size_t)4 * 2048 * 1024;  // elems per bf16 scratch tensor
  bf16* qws = (bf16*)d_ws;
  bf16* kws = qws + T;
  bf16* vws = kws + T;
  bf16* ows = vws + T;   // attention output, [B*N, C] bf16

  dim3 blk(256);
  (gemm_bf16_wmma<float, 0>)<<<dim3(3072 / BN, 8192 / BM), blk, 0, stream>>>(
      x, qkv_w, qkv_b, nullptr, qws, kws, vws, 8192, 3072, 1024);
  attn_flash_wmma<<<dim3(2048 / 128, 64), blk, 0, stream>>>(qws, kws, vws, ows);
  (gemm_bf16_wmma<bf16, 1>)<<<dim3(1024 / BN, 8192 / BM), blk, 0, stream>>>(
      ows, fc_w, fc_b, out, nullptr, nullptr, nullptr, 8192, 1024, 1024);
}
